// WindowAttention_2551210574518
// MI455X (gfx1250) — matrix operand order repaired and checked by reading, not compiled, not run on hardware
//
#include <hip/hip_runtime.h>
#include <hip/hip_bf16.h>

typedef _Float16 f16;
typedef __attribute__((ext_vector_type(16))) _Float16 v16h;
typedef __attribute__((ext_vector_type(8)))  float    v8f;

#define NTOK   49
#define NPAD   64
#define HEADS  8
#define HDIM   32
#define DIM    256
#define QKVDIM 768
#define NW     64
#define NWIN   4096

// ---------------- workspace layout (bytes) ----------------
// [0, 393216)        : w_qkv^T as f16  [768][256]   (col-major of original)
// [393216, 524288)   : w_out^T as f16  [256][256]
// [524288, 601120)   : bias matrix f32 [8][49][49]
#define WS_WQKVT 0
#define WS_WOUTT 393216
#define WS_BIAS  524288

__device__ __forceinline__ v8f vzero8() {
  v8f z = {0.f,0.f,0.f,0.f,0.f,0.f,0.f,0.f};
  return z;
}

// A-matrix fragment (16x32 f16), storage row-major [M][K], stride in halves.
// ISA layout: lane(0..15)=M rows, halves: VGPR0..3 -> K = h*8 + 0..7,
// VGPR4..7 -> K = 16 + h*8 + 0..7 (h = lane>>4).
__device__ __forceinline__ v16h load_fragA(const f16* p, int stride, int kbase, int lane) {
  int r  = lane & 15;
  int hh = lane >> 4;
  const f16* b = p + r * stride + kbase + hh * 8;
  v16h out;
  ((uint4*)&out)[0] = *(const uint4*)(b);       // K = kbase + hh*8 + 0..7
  ((uint4*)&out)[1] = *(const uint4*)(b + 16);  // K = kbase + 16 + hh*8 + 0..7
  return out;
}

// B-matrix fragment (32x16 f16), storage N-major [N][K] (i.e. transposed),
// stride in halves. ISA layout: lane half h covers K = h*8 + 0..7 and 16 + h*8 + 0..7, N = lane&15.
__device__ __forceinline__ v16h load_fragB(const f16* p, int stride, int kbase, int lane) {
  int n  = lane & 15;
  int hh = lane >> 4;
  const f16* b = p + n * stride + kbase + hh * 8;
  v16h out;
  ((uint4*)&out)[0] = *(const uint4*)(b);       // K = kbase + hh*8 + 0..7
  ((uint4*)&out)[1] = *(const uint4*)(b + 16);  // K = kbase + 16 + hh*8 + 0..7
  return out;
}

__device__ __forceinline__ v8f wmma_f16(v16h a, v16h b, v8f c) {
  return __builtin_amdgcn_wmma_f32_16x16x32_f16(false, a, false, b, (short)0, c,
                                                false, false);
}

// ---------------- prep: transpose/convert weights, build bias ----------------
__global__ void swin_prep_kernel(const float* __restrict__ w_qkv,
                                 const float* __restrict__ w_out,
                                 const float* __restrict__ rel_table,
                                 f16* __restrict__ wqkvT,
                                 f16* __restrict__ woutT,
                                 float* __restrict__ biasM) {
  int stride = gridDim.x * blockDim.x;
  int tid = blockIdx.x * blockDim.x + threadIdx.x;
  for (int i = tid; i < QKVDIM * DIM; i += stride) {   // wqkvT[n][k] = w_qkv[k][n]
    int n = i >> 8, k = i & 255;
    wqkvT[i] = (f16)w_qkv[k * QKVDIM + n];
  }
  for (int i = tid; i < DIM * DIM; i += stride) {      // woutT[n][k] = w_out[k][n]
    int n = i >> 8, k = i & 255;
    woutT[i] = (f16)w_out[k * DIM + n];
  }
  for (int i = tid; i < HEADS * NTOK * NTOK; i += stride) {
    int h = i / (NTOK * NTOK);
    int rest = i % (NTOK * NTOK);
    int ii = rest / NTOK, jj = rest % NTOK;
    int rh = (ii / 7) - (jj / 7) + 6;
    int rw = (ii % 7) - (jj % 7) + 6;
    int idx = rh * 13 + rw;
    biasM[i] = rel_table[idx * HEADS + h];
  }
}

// ---------------- main: one window per workgroup (8 waves) ----------------
// Dynamic LDS (128 KB):
//   [0,      32768)  Qs [8][64][32] f16   (later overlaid by P [8][64][64] f16)
//   [32768,  65536)  Ks [8][64][32] f16   (second half of P overlay)
//   [65536,  98304)  Vt [8][32][64] f16   (V stored transposed: [d][token])
//   [98304, 131072)  xh [64][256]  f16    (later reused as Y [64][256] f16)
__global__ void swin_attn_kernel(const float* __restrict__ x,
                                 const float* __restrict__ mask,
                                 const float* __restrict__ b_qkv,
                                 const float* __restrict__ b_out,
                                 const f16*   __restrict__ wqkvT,
                                 const f16*   __restrict__ woutT,
                                 const float* __restrict__ biasM,
                                 float* __restrict__ out) {
  extern __shared__ __align__(16) char smem[];
  f16* Pp = (f16*)smem;                     // [8][64][64]
  f16* Qs = (f16*)smem;                     // [8][64][32]
  f16* Ks = (f16*)(smem + 32768);           // [8][64][32]
  f16* Vt = (f16*)(smem + 65536);           // [8][32][64]
  f16* xh = (f16*)(smem + 98304);           // [64][256], reused as Y

  const int win  = blockIdx.x;
  const int nw   = win & (NW - 1);
  const int tid  = threadIdx.x;
  const int lane = tid & 31;
  const int wv   = tid >> 5;                // wave id 0..7
  const int l15  = lane & 15;
  const int hh2  = lane >> 4;

  // ---- stage 1: stage x window into LDS as f16, zero-pad rows 49..63 ----
  const float* xw = x + (size_t)win * NTOK * DIM;
  for (int idx = tid; idx < NTOK * DIM; idx += 256) xh[idx] = (f16)xw[idx];
  for (int idx = NTOK * DIM + tid; idx < NPAD * DIM; idx += 256) xh[idx] = (f16)0.f;
  __syncthreads();

  // ---- stage 2: QKV = x @ w_qkv + b  (M=64pad, N=768, K=256) ----
  // wave wv owns N-tiles [6*wv, 6*wv+6)
  for (int g = 0; g < 6; ++g) {
    const int nt = wv * 6 + g;
    const int colbase = nt * 16;
    const int col = colbase + l15;
    const float bq = b_qkv[col];
    v8f acc[4];
#pragma unroll
    for (int mt = 0; mt < 4; ++mt) acc[mt] = vzero8();
    for (int ks = 0; ks < 8; ++ks) {
      v16h bf = load_fragB(wqkvT + colbase * DIM, DIM, ks * 32, lane);
#pragma unroll
      for (int mt = 0; mt < 4; ++mt) {
        v16h af = load_fragA(xh + mt * 16 * DIM, DIM, ks * 32, lane);
        acc[mt] = wmma_f16(af, bf, acc[mt]);
      }
    }
    const int qi = col >> 8;            // 0=Q,1=K,2=V
    const int h8 = (col >> 5) & 7;      // head
    const int dd = col & 31;            // dim within head
#pragma unroll
    for (int mt = 0; mt < 4; ++mt) {
#pragma unroll
      for (int r = 0; r < 8; ++r) {
        const int row = mt * 16 + r + hh2 * 8;
        const f16 v = (f16)(acc[mt][r] + bq);
        if (qi == 0)      Qs[(h8 * 64 + row) * 32 + dd] = v;
        else if (qi == 1) Ks[(h8 * 64 + row) * 32 + dd] = v;
        else              Vt[(h8 * 32 + dd) * 64 + row] = v;   // transposed
      }
    }
  }
  __syncthreads();

  // ---- stage 3: per-head attention; wave wv owns head wv ----
  const int h = wv;
  const f16* Kh = Ks + h * 64 * 32;
  const f16* Qh = Qs + h * 64 * 32;
  const f16* Vh = Vt + h * 32 * 64;

  // S = K @ Q^T  (M=i, N=j, K=32) -> 4x4 tiles, one K-step
  v8f s[4][4];
#pragma unroll
  for (int nt = 0; nt < 4; ++nt) {
    v16h bf = load_fragB(Qh + nt * 16 * 32, 32, 0, lane);
#pragma unroll
    for (int mt = 0; mt < 4; ++mt) {
      v16h af = load_fragA(Kh + mt * 16 * 32, 32, 0, lane);
      s[nt][mt] = wmma_f16(af, bf, vzero8());
    }
  }
  __syncthreads();   // all waves done reading Q/K before P overlays them

  // softmax over j (last axis), scale + bias + mask, write P (f16) to LDS
  const float* biasH = biasM + h * NTOK * NTOK;
  const float* maskW = mask + (size_t)nw * NTOK * NTOK;
#pragma unroll
  for (int mt = 0; mt < 4; ++mt) {
#pragma unroll
    for (int r = 0; r < 8; ++r) {
      const int i = mt * 16 + r + hh2 * 8;
      float sv[4];
#pragma unroll
      for (int nt = 0; nt < 4; ++nt) {
        const int j = nt * 16 + l15;
        float t = s[nt][mt][r] * 0.17677669529663689f;  // 1/sqrt(32)
        if (i < NTOK && j < NTOK) t += biasH[i * NTOK + j] + maskW[i * NTOK + j];
        if (j >= NTOK) t = -1e30f;
        sv[nt] = t;
      }
      float m = fmaxf(fmaxf(sv[0], sv[1]), fmaxf(sv[2], sv[3]));
      m = fmaxf(m, __shfl_xor(m, 1, 16));
      m = fmaxf(m, __shfl_xor(m, 2, 16));
      m = fmaxf(m, __shfl_xor(m, 4, 16));
      m = fmaxf(m, __shfl_xor(m, 8, 16));
      float sum = 0.f;
#pragma unroll
      for (int nt = 0; nt < 4; ++nt) { sv[nt] = __expf(sv[nt] - m); sum += sv[nt]; }
      sum += __shfl_xor(sum, 1, 16);
      sum += __shfl_xor(sum, 2, 16);
      sum += __shfl_xor(sum, 4, 16);
      sum += __shfl_xor(sum, 8, 16);
      const float inv = 1.f / sum;
#pragma unroll
      for (int nt = 0; nt < 4; ++nt)
        Pp[h * 64 * 64 + i * 64 + nt * 16 + l15] = (f16)(sv[nt] * inv);
    }
  }
  __syncthreads();

  // Y = P @ V  (M=i, N=d=32, K=64 -> 2 K-steps); store into xh region (x dead)
#pragma unroll
  for (int ntd = 0; ntd < 2; ++ntd) {
#pragma unroll
    for (int mt = 0; mt < 4; ++mt) {
      v8f acy = vzero8();
#pragma unroll
      for (int ks = 0; ks < 2; ++ks) {
        v16h bf = load_fragB(Vh + ntd * 16 * 64, 64, ks * 32, lane);
        v16h af = load_fragA(Pp + h * 64 * 64 + mt * 16 * 64, 64, ks * 32, lane);
        acy = wmma_f16(af, bf, acy);
      }
#pragma unroll
      for (int r = 0; r < 8; ++r) {
        const int row = mt * 16 + r + hh2 * 8;
        xh[row * DIM + h * 32 + ntd * 16 + l15] = (f16)acy[r];
      }
    }
  }
  __syncthreads();

  // ---- stage 4: out = Y @ w_out + b_out  (M=64pad, N=256, K=256) ----
  float* outW = out + (size_t)win * NTOK * DIM;
  for (int g = 0; g < 2; ++g) {
    const int nt = wv * 2 + g;
    const int colbase = nt * 16;
    const int col = colbase + l15;
    const float bo = b_out[col];
    v8f acc[4];
#pragma unroll
    for (int mt = 0; mt < 4; ++mt) acc[mt] = vzero8();
    for (int ks = 0; ks < 8; ++ks) {
      v16h bf = load_fragB(woutT + colbase * DIM, DIM, ks * 32, lane);
#pragma unroll
      for (int mt = 0; mt < 4; ++mt) {
        v16h af = load_fragA(xh + mt * 16 * DIM, DIM, ks * 32, lane);
        acc[mt] = wmma_f16(af, bf, acc[mt]);
      }
    }
#pragma unroll
    for (int mt = 0; mt < 4; ++mt) {
#pragma unroll
      for (int r = 0; r < 8; ++r) {
        const int row = mt * 16 + r + hh2 * 8;
        if (row < NTOK) outW[row * DIM + col] = acc[mt][r] + bo;
      }
    }
  }
}

extern "C" void kernel_launch(void* const* d_in, const int* in_sizes, int n_in,
                              void* d_out, int out_size, void* d_ws, size_t ws_size,
                              hipStream_t stream) {
  const float* x         = (const float*)d_in[0];
  const float* mask      = (const float*)d_in[1];
  const float* w_qkv     = (const float*)d_in[2];
  const float* b_qkv     = (const float*)d_in[3];
  const float* rel_table = (const float*)d_in[4];
  const float* w_out     = (const float*)d_in[5];
  const float* b_out     = (const float*)d_in[6];

  f16*   wqkvT = (f16*)((char*)d_ws + WS_WQKVT);
  f16*   woutT = (f16*)((char*)d_ws + WS_WOUTT);
  float* biasM = (float*)((char*)d_ws + WS_BIAS);

  swin_prep_kernel<<<dim3(208), dim3(256), 0, stream>>>(
      w_qkv, w_out, rel_table, wqkvT, woutT, biasM);

  swin_attn_kernel<<<dim3(NWIN), dim3(256), 131072, stream>>>(
      x, mask, b_qkv, b_out, wqkvT, woutT, biasM, (float*)d_out);
}
